// DualHelixAttentionBlock_22935125360705
// MI455X (gfx1250) — matrix operand order repaired and checked by reading, not compiled, not run on hardware
//
#include <hip/hip_runtime.h>
#include <math.h>

// ---------------------------------------------------------------------------
// Types for WMMA (gfx1250, wave32)
// ---------------------------------------------------------------------------
typedef __attribute__((ext_vector_type(16))) __bf16 v16bf;
typedef __attribute__((ext_vector_type(8)))  float  v8f;
typedef int v4i_gcc __attribute__((vector_size(16)));   // matches builtin's V4i

#define AS_GLOBAL __attribute__((address_space(1)))
#define AS_LOCAL  __attribute__((address_space(3)))

#if defined(__AMDGCN__) && __has_builtin(__builtin_amdgcn_global_load_async_to_lds_b128) && __has_builtin(__builtin_amdgcn_s_wait_asynccnt)
#define USE_ASYNC_LDS 1
#else
#define USE_ASYNC_LDS 0
#endif

__device__ __forceinline__ float gelu_exact(float x) {
    return 0.5f * x * (1.0f + erff(x * 0.70710678118654752f));
}

// ---------------------------------------------------------------------------
// WMMA GEMM:  Y[M,N] = act( X[M,K] * W[N,K]^T + bias[N] + Res[M,N] )
//   X row stride = lda, Y row stride = ldy, Res row stride = ldr.
//   M,N multiples of 64; K multiple of 32.  fp32 in, bf16 WMMA, fp32 out.
// Block = 128 threads (4 waves); block tile 64x64; wave tile 32x32 (2x2 WMMA).
// ---------------------------------------------------------------------------
#define GEMM_LDSS 36  // 32 + 4 pad floats per LDS row (144B, 16B aligned)

__global__ void __launch_bounds__(128)
k_gemm_wmma(const float* __restrict__ X, int lda,
            const float* __restrict__ Wt,        // [N,K] row-major (o,i)
            const float* __restrict__ bias,      // [N] or null
            const float* __restrict__ Res, int ldr,
            float* __restrict__ Y, int ldy,
            int N, int K, int act)
{
    __shared__ float As[64 * GEMM_LDSS];
    __shared__ float Bs[64 * GEMM_LDSS];

    const int tid  = threadIdx.x;
    const int lane = tid & 31;
    const int wave = tid >> 5;
    const int wm   = (wave >> 1) * 32;
    const int wn   = (wave & 1) * 32;
    const int m0   = blockIdx.y * 64;
    const int n0   = blockIdx.x * 64;
    const int half = lane >> 4;
    const int l16  = lane & 15;

    v8f zero = {};
    v8f acc[2][2];
    #pragma unroll
    for (int i = 0; i < 2; ++i)
        #pragma unroll
        for (int j = 0; j < 2; ++j) acc[i][j] = zero;

    for (int k0 = 0; k0 < K; k0 += 32) {
        // Stage A (64x32) and B (64x32) fp32 tiles into LDS. 2048 floats each,
        // 128 threads x 4 float4.
        #pragma unroll
        for (int j = 0; j < 4; ++j) {
            const int f   = (tid + j * 128) << 2;   // 0..2044 step 4
            const int row = f >> 5;
            const int col = f & 31;
            const float* ga = X  + (size_t)(m0 + row) * lda + (k0 + col);
            const float* gb = Wt + (size_t)(n0 + row) * K   + (k0 + col);
#if USE_ASYNC_LDS
            __builtin_amdgcn_global_load_async_to_lds_b128(
                (AS_GLOBAL v4i_gcc*)ga,
                (AS_LOCAL  v4i_gcc*)&As[row * GEMM_LDSS + col], 0, 0);
            __builtin_amdgcn_global_load_async_to_lds_b128(
                (AS_GLOBAL v4i_gcc*)gb,
                (AS_LOCAL  v4i_gcc*)&Bs[row * GEMM_LDSS + col], 0, 0);
#else
            *(float4*)&As[row * GEMM_LDSS + col] = *(const float4*)ga;
            *(float4*)&Bs[row * GEMM_LDSS + col] = *(const float4*)gb;
#endif
        }
        if (k0 + 32 < K) {
            // hint the next A tile toward L2 (lowers to global_prefetch_b8)
            __builtin_prefetch(X + (size_t)(m0 + (tid & 63)) * lda + (k0 + 32), 0, 1);
        }
#if USE_ASYNC_LDS
        __builtin_amdgcn_s_wait_asynccnt(0);
#endif
        __syncthreads();

        // Build bf16 fragments from the fp32 LDS tiles.
        // A (16x32): lane holds row l16; K chunks [8*half, 8*half+8) and +16.
        // B (32x16): lane holds col l16; K chunks [8*half, 8*half+8) and +16.
        v16bf af[2], bfr[2];
        #pragma unroll
        for (int i = 0; i < 2; ++i) {
            const float* ar = &As[(wm + i * 16 + l16) * GEMM_LDSS];
            const int kb = half * 8;
            #pragma unroll
            for (int e = 0; e < 8; ++e) af[i][e]     = (__bf16)ar[kb + e];
            #pragma unroll
            for (int e = 0; e < 8; ++e) af[i][8 + e] = (__bf16)ar[16 + kb + e];
        }
        #pragma unroll
        for (int j = 0; j < 2; ++j) {
            const float* br = &Bs[(wn + j * 16 + l16) * GEMM_LDSS];
            const int kb = half * 8;
            #pragma unroll
            for (int e = 0; e < 8; ++e) bfr[j][e]     = (__bf16)br[kb + e];
            #pragma unroll
            for (int e = 0; e < 8; ++e) bfr[j][8 + e] = (__bf16)br[16 + kb + e];
        }

        #pragma unroll
        for (int i = 0; i < 2; ++i)
            #pragma unroll
            for (int j = 0; j < 2; ++j)
                acc[i][j] = __builtin_amdgcn_wmma_f32_16x16x32_bf16(
                    false, af[i], false, bfr[j], (short)0, acc[i][j], false, false);

        __syncthreads();
    }

    // Epilogue. C/D layout: VGPR rr, lane -> (M = rr + 8*half, N = l16).
    #pragma unroll
    for (int i = 0; i < 2; ++i) {
        #pragma unroll
        for (int j = 0; j < 2; ++j) {
            const int colg = n0 + wn + j * 16 + l16;
            const float bv = bias ? bias[colg] : 0.0f;
            #pragma unroll
            for (int rr = 0; rr < 8; ++rr) {
                const int rowg = m0 + wm + i * 16 + half * 8 + rr;
                float v = acc[i][j][rr] + bv;
                if (Res) v += Res[(size_t)rowg * ldr + colg];
                if (act == 1) v = gelu_exact(v);
                Y[(size_t)rowg * ldy + colg] = v;
            }
        }
    }
}

// ---------------------------------------------------------------------------
// Layout helpers
// ---------------------------------------------------------------------------
__global__ void k_nchw_to_tok(const float* __restrict__ X, float* __restrict__ Y,
                              int B, int C, int H, int W)
{
    size_t i = (size_t)blockIdx.x * blockDim.x + threadIdx.x;
    size_t total = (size_t)B * C * H * W;
    if (i >= total) return;
    int c = (int)(i % C);
    size_t tok = i / C;
    int w = (int)(tok % W);
    size_t t2 = tok / W;
    int h = (int)(t2 % H);
    int b = (int)(t2 / H);
    Y[i] = X[(((size_t)b * C + c) * H + h) * W + w];
}

__global__ void k_tok_to_nchw(const float* __restrict__ X, float* __restrict__ Y,
                              int B, int C, int H, int W)
{
    size_t i = (size_t)blockIdx.x * blockDim.x + threadIdx.x;
    size_t total = (size_t)B * C * H * W;
    if (i >= total) return;
    // i indexes NCHW output
    int w = (int)(i % W);
    size_t t2 = i / W;
    int h = (int)(t2 % H);
    size_t t3 = t2 / H;
    int c = (int)(t3 % C);
    int b = (int)(t3 / C);
    Y[i] = X[(((size_t)b * H + h) * W + w) * C + c];
}

__global__ void k_split(const float* __restrict__ XE, float* __restrict__ S,
                        float* __restrict__ F, size_t M)
{
    size_t i = (size_t)blockIdx.x * blockDim.x + threadIdx.x;
    if (i >= M * 128) return;
    size_t m = i >> 7;
    int c = (int)(i & 127);
    S[i] = XE[(m << 8) + c];
    F[i] = XE[(m << 8) + 128 + c];
}

__global__ void k_concat_add(const float* __restrict__ XE, const float* __restrict__ S,
                             const float* __restrict__ F, float* __restrict__ R, size_t M)
{
    size_t i = (size_t)blockIdx.x * blockDim.x + threadIdx.x;
    if (i >= M * 256) return;
    size_t m = i >> 8;
    int c = (int)(i & 255);
    float add = (c < 128) ? S[(m << 7) + c] : F[(m << 7) + (c - 128)];
    R[i] = XE[i] + add;
}

// ---------------------------------------------------------------------------
// GroupNorm over whole sample (mean/var across C*H*W).  Token-major makes the
// per-sample region contiguous.  One block per batch sample.
// ---------------------------------------------------------------------------
__global__ void __launch_bounds__(256)
k_groupnorm(const float* __restrict__ X, const float* __restrict__ g,
            const float* __restrict__ be, float* __restrict__ Y, int C, size_t n)
{
    __shared__ float red[512];
    const int b = blockIdx.x;
    const int tid = threadIdx.x;
    const float* xp = X + (size_t)b * n;
    float s = 0.f, s2 = 0.f;
    for (size_t i = tid; i < n; i += 256) { float v = xp[i]; s += v; s2 += v * v; }
    red[tid] = s; red[256 + tid] = s2;
    __syncthreads();
    for (int off = 128; off > 0; off >>= 1) {
        if (tid < off) { red[tid] += red[tid + off]; red[256 + tid] += red[256 + tid + off]; }
        __syncthreads();
    }
    const float mean = red[0] / (float)n;
    const float var  = red[256] / (float)n - mean * mean;
    const float rs   = rsqrtf(var + 1e-5f);
    float* yp = Y + (size_t)b * n;
    for (size_t i = tid; i < n; i += 256) {
        int c = (int)(i % (size_t)C);
        yp[i] = (xp[i] - mean) * rs * g[c] + be[c];
    }
}

// ---------------------------------------------------------------------------
// conv3x3 (SAME) + GELU, token-major.  One block per token, one thread per
// output channel (C = 128).
// ---------------------------------------------------------------------------
__global__ void __launch_bounds__(128)
k_conv3x3_gelu(const float* __restrict__ X, const float* __restrict__ Wc,
               const float* __restrict__ bias, float* __restrict__ Y,
               int B, int H, int W, int C)
{
    __shared__ float xin[9 * 128];
    const int tok = blockIdx.x;
    const int w = tok % W;
    const int t2 = tok / W;
    const int h = t2 % H;
    const int b = t2 / H;
    const int tid = threadIdx.x;
    for (int e = tid; e < 9 * C; e += blockDim.x) {
        int nn = e / C, c = e - nn * C;
        int hh = h + nn / 3 - 1, ww = w + nn % 3 - 1;
        xin[e] = (hh >= 0 && hh < H && ww >= 0 && ww < W)
                     ? X[(((size_t)b * H + hh) * W + ww) * C + c] : 0.0f;
    }
    __syncthreads();
    const int o = tid;
    float acc = bias[o];
    for (int nn = 0; nn < 9; ++nn) {
        const int dy = nn / 3, dx = nn % 3;
        const float* wr = Wc + (size_t)o * C * 9 + dy * 3 + dx;
        const float* xr = &xin[nn * C];
        for (int c = 0; c < C; ++c) acc += xr[c] * wr[(size_t)c * 9];
    }
    Y[(size_t)tok * C + o] = gelu_exact(acc);
}

// ---------------------------------------------------------------------------
// DCT-II / inverse, 64x64 per (b, c).  Y = D X D^T (fwd) / D^T Y D (inv).
// ---------------------------------------------------------------------------
__global__ void k_dct_init(float* D)
{
    int e = blockIdx.x * blockDim.x + threadIdx.x;
    if (e >= 4096) return;
    int k = e >> 6, n = e & 63;
    float m = cosf(3.14159265358979323846f * (2.0f * n + 1.0f) * (float)k / 128.0f);
    float s = (k == 0) ? 0.125f : 0.17677669529663687f; // sqrt(1/64), sqrt(2/64)
    D[e] = m * s;
}

__global__ void __launch_bounds__(256)
k_dct2d(const float* __restrict__ X, float* __restrict__ Y,
        const float* __restrict__ D, int C, int inverse)
{
    __shared__ float Xs[4096];
    __shared__ float Ts[4096];
    const int b = blockIdx.x / C;
    const int c = blockIdx.x % C;
    const int tid = threadIdx.x;
    const size_t base = (size_t)b * 4096;
    for (int e = tid; e < 4096; e += 256) Xs[e] = X[(base + e) * C + c];
    __syncthreads();
    for (int e = tid; e < 4096; e += 256) {
        const int i = e >> 6, j = e & 63;
        float a = 0.f;
        if (!inverse) { for (int h = 0; h < 64; ++h) a += D[i * 64 + h] * Xs[h * 64 + j]; }
        else          { for (int h = 0; h < 64; ++h) a += D[h * 64 + i] * Xs[h * 64 + j]; }
        Ts[e] = a;
    }
    __syncthreads();
    for (int e = tid; e < 4096; e += 256) {
        const int i = e >> 6, j = e & 63;
        float a = 0.f;
        if (!inverse) { for (int s = 0; s < 64; ++s) a += Ts[i * 64 + s] * D[j * 64 + s]; }
        else          { for (int s = 0; s < 64; ++s) a += Ts[i * 64 + s] * D[s * 64 + j]; }
        Y[(base + e) * C + c] = a;
    }
}

// ---------------------------------------------------------------------------
// Unified windowed MHA: dense (d=1,s=0), shifted (s>0), dilated sparse (d>1).
// Zero padding of Q/K/V beyond HxW exactly as the reference (padded keys
// participate in softmax; padded queries are dropped).  Accumulates
// O += scale * attn  (O must be pre-zeroed per instance group).
// nh=4, dh=32 fixed.  One block per (b, sub, window, head).
// ---------------------------------------------------------------------------
#define ADH 32
#define ALMAX 121

__global__ void __launch_bounds__(128)
k_win_attn(const float* __restrict__ Q, const float* __restrict__ Kp,
           const float* __restrict__ Vp, float* __restrict__ O,
           int B, int H, int W, int C, int nh,
           int r, int d, int shift, int Hp, int Wp, float scale)
{
    __shared__ float Ks[ALMAX * ADH];
    __shared__ float Vs[ALMAX * ADH];
    const int Hs = Hp / d, Ws = Wp / d;
    const int wx = Ws / r;
    const int L = r * r;
    const int nwin = (Hs / r) * wx;
    const int nsub = d * d;
    int idx = blockIdx.x;
    const int head = idx % nh; idx /= nh;
    const int win = idx % nwin; idx /= nwin;
    const int sub = idx % nsub; idx /= nsub;
    const int b = idx;
    const int dy = sub / d, dx = sub % d;
    const int wy0 = (win / wx) * r, wx0 = (win % wx) * r;
    const int c0 = head * ADH;
    const int tid = threadIdx.x;

    for (int e = tid; e < L * ADH; e += blockDim.x) {
        const int p = e / ADH, c = e - p * ADH;
        const int hh = wy0 + p / r, ww = wx0 + p % r;
        const int ph = hh * d + dy, pw = ww * d + dx;
        float kv = 0.f, vv = 0.f;
        if (ph < H && pw < W) {
            int oh = ph + shift; if (oh >= H) oh -= H;
            int ow = pw + shift; if (ow >= W) ow -= W;
            const size_t t = ((size_t)(b * H + oh) * W + ow) * C + c0 + c;
            kv = Kp[t]; vv = Vp[t];
        }
        Ks[e] = kv; Vs[e] = vv;
    }
    __syncthreads();
    if (tid >= L) return;

    const int p = tid;
    const int hh = wy0 + p / r, ww = wx0 + p % r;
    const int ph = hh * d + dy, pw = ww * d + dx;
    const bool qvalid = (ph < H && pw < W);
    size_t qbase = 0;
    float q[ADH];
    if (qvalid) {
        int oh = ph + shift; if (oh >= H) oh -= H;
        int ow = pw + shift; if (ow >= W) ow -= W;
        qbase = ((size_t)(b * H + oh) * W + ow) * C + c0;
        #pragma unroll
        for (int c = 0; c < ADH; ++c) q[c] = Q[qbase + c];
    } else {
        #pragma unroll
        for (int c = 0; c < ADH; ++c) q[c] = 0.f;
    }

    float m = -3.0e38f, s = 0.f, o[ADH];
    #pragma unroll
    for (int c = 0; c < ADH; ++c) o[c] = 0.f;
    const float sc = 0.17677669529663687f; // 1/sqrt(32)
    for (int j = 0; j < L; ++j) {
        float l = 0.f;
        #pragma unroll
        for (int c = 0; c < ADH; ++c) l += q[c] * Ks[j * ADH + c];
        l *= sc;
        const float nm = l > m ? l : m;
        const float corr = __expf(m - nm);
        const float pj = __expf(l - nm);
        s = s * corr + pj;
        #pragma unroll
        for (int c = 0; c < ADH; ++c) o[c] = o[c] * corr + pj * Vs[j * ADH + c];
        m = nm;
    }
    if (qvalid) {
        const float inv = scale / s;
        #pragma unroll
        for (int c = 0; c < ADH; ++c) O[qbase + c] += o[c] * inv;
    }
}

// ---------------------------------------------------------------------------
// Channel cross-attention: per (b, head, chunk-of-64-tokens), attention over
// channels.  cs = 64 fixed.  One block per (b, head, chunk); thread = channel.
// ---------------------------------------------------------------------------
#define CCS 64
#define CCHMAX 128

__global__ void __launch_bounds__(128)
k_chan_attn(const float* __restrict__ Q, const float* __restrict__ Kp,
            const float* __restrict__ Vp, float* __restrict__ O,
            int B, int C, int HW, int nh)
{
    __shared__ float Ks[CCHMAX * CCS];
    __shared__ float Vs[CCHMAX * CCS];
    const int ch = C / nh;
    const int nch = HW / CCS;
    int idx = blockIdx.x;
    const int n = idx % nch; idx /= nch;
    const int h = idx % nh; idx /= nh;
    const int b = idx;
    const int cbase = h * ch;
    const int l0 = n * CCS;
    const int tid = threadIdx.x;

    for (int e = tid; e < ch * CCS; e += blockDim.x) {
        const int c = e / CCS, t = e - c * CCS;
        const size_t g = ((size_t)(b * HW) + l0 + t) * C + cbase + c;
        Ks[e] = Kp[g]; Vs[e] = Vp[g];
    }
    __syncthreads();
    if (tid >= ch) return;

    float q[CCS];
    for (int t = 0; t < CCS; ++t)
        q[t] = Q[((size_t)(b * HW) + l0 + t) * C + cbase + tid];

    float m = -3.0e38f, s = 0.f, o[CCS];
    for (int t = 0; t < CCS; ++t) o[t] = 0.f;
    const float sc = 0.125f; // 1/sqrt(64)
    for (int dc = 0; dc < ch; ++dc) {
        float l = 0.f;
        for (int t = 0; t < CCS; ++t) l += q[t] * Ks[dc * CCS + t];
        l *= sc;
        const float nm = l > m ? l : m;
        const float corr = __expf(m - nm);
        const float pj = __expf(l - nm);
        s = s * corr + pj;
        for (int t = 0; t < CCS; ++t) o[t] = o[t] * corr + pj * Vs[dc * CCS + t];
        m = nm;
    }
    const float inv = 1.f / s;
    for (int t = 0; t < CCS; ++t)
        O[((size_t)(b * HW) + l0 + t) * C + cbase + tid] = o[t] * inv;
}

// ---------------------------------------------------------------------------
// Host orchestration
// ---------------------------------------------------------------------------
enum {
    IN_X = 0,
    P_PROJ_W = 1, P_PROJ_B = 2,
    P_SIC_W = 3, P_SIC_B = 4,
    P_SINTER_W = 5, P_SINTER_B = 6,
    P_FIC_W = 7, P_FIC_B = 8,
    P_FINTER_W = 9, P_FINTER_B = 10,
    P_SN1_G = 11, P_SN1_B = 12, P_SN2_G = 13, P_SN2_B = 14,
    P_SN3_G = 15, P_SN3_B = 16, P_SN4_G = 17, P_SN4_B = 18,
    P_FN1_G = 19, P_FN1_B = 20, P_FN2_G = 21, P_FN2_B = 22,
    P_FN3_G = 23, P_FN3_B = 24, P_FN4_G = 25, P_FN4_B = 26,
    P_SDA = 27, P_FDA = 35, P_SCA1 = 43, P_FCA1 = 51,
    P_SSA = 59, P_FSA = 67, P_SCA2 = 75, P_FCA2 = 83,
    P_SFFN1 = 91, P_SFFN2 = 95, P_SFFN3 = 99, P_SFFN4 = 103,
    P_FFFN1 = 107, P_FFFN2 = 111, P_FFFN3 = 115, P_FFFN4 = 119,
    P_OUT_W = 123, P_OUT_B = 124, P_OUTN_G = 125, P_OUTN_B = 126
};

extern "C" void kernel_launch(void* const* d_in, const int* in_sizes, int n_in,
                              void* d_out, int out_size, void* d_ws, size_t ws_size,
                              hipStream_t stream)
{
    (void)in_sizes; (void)n_in; (void)out_size; (void)ws_size;
    const int B = 8, H = 64, W = 64, C = 128;
    const size_t M = (size_t)B * H * W;   // 32768 tokens
    const size_t szC = M * 128;           // elements of one [M,128] buffer

    const float* P[127];
    for (int i = 0; i < 127; ++i) P[i] = (const float*)d_in[i];

    // Carve workspace
    size_t off = 0;
    auto carve = [&](size_t floats) -> float* {
        float* p = (float*)((char*)d_ws + off);
        off += ((floats * sizeof(float)) + 255) & ~(size_t)255;
        return p;
    };
    float* TX  = carve(M * 128);
    float* XE  = carve(M * 256);
    float* SA  = carve(M * 128);
    float* SB  = carve(M * 128);
    float* SC  = carve(M * 128);
    float* FA  = carve(M * 128);
    float* FB  = carve(M * 128);
    float* FC  = carve(M * 128);
    float* Qb  = carve(M * 128);
    float* Kb  = carve(M * 128);
    float* Vb  = carve(M * 128);
    float* ACC = carve(M * 128);
    float* Hb  = carve(M * 512);
    float* RSB = carve(M * 256);
    float* Dm  = carve(4096);

    auto gemm = [&](const float* X, int lda, const float* Wt, const float* bias,
                    const float* Res, int ldr, float* Y, int ldy, int N, int K, int act) {
        dim3 g(N / 64, (unsigned)(M / 64));
        k_gemm_wmma<<<g, 128, 0, stream>>>(X, lda, Wt, bias, Res, ldr, Y, ldy, N, K, act);
    };
    auto gnorm = [&](const float* x, const float* g, const float* b, float* y) {
        k_groupnorm<<<B, 256, 0, stream>>>(x, g, b, y, C, (size_t)H * W * C);
    };
    auto winattn = [&](const float* q, const float* k, const float* v, float* o,
                       int r, int d, int shift, float scale) {
        const int m = r * d;
        const int Hp = ((H + m - 1) / m) * m;
        const int Hs = Hp / d;
        const int nwin = (Hs / r) * (Hs / r);
        const int grid = B * d * d * nwin * 4;
        k_win_attn<<<grid, 128, 0, stream>>>(q, k, v, o, B, H, W, C, 4, r, d, shift, Hp, Hp, scale);
    };
    auto qkv = [&](const float* qsrc, const float* kvsrc, int pb) {
        gemm(qsrc, 128, P[pb + 0], P[pb + 1], nullptr, 0, Qb, 128, 128, 128, 0);
        gemm(kvsrc, 128, P[pb + 2], P[pb + 3], nullptr, 0, Kb, 128, 128, 128, 0);
        gemm(kvsrc, 128, P[pb + 4], P[pb + 5], nullptr, 0, Vb, 128, 128, 128, 0);
    };
    auto ffn = [&](const float* x, int pb, float* out) {
        gemm(x, 128, P[pb + 0], P[pb + 1], nullptr, 0, Hb, 512, 512, 128, 1);
        gemm(Hb, 512, P[pb + 2], P[pb + 3], x, 128, out, 128, 128, 512, 0);
    };
    auto zero_acc = [&]() { (void)hipMemsetAsync(ACC, 0, szC * sizeof(float), stream); };

    // ---- forward ----
    k_dct_init<<<16, 256, 0, stream>>>(Dm);
    {
        const size_t total = (size_t)B * C * H * W;
        k_nchw_to_tok<<<(unsigned)((total + 255) / 256), 256, 0, stream>>>(P[IN_X], TX, B, C, H, W);
    }
    // xe = proj(x)  [M,256]
    gemm(TX, 128, P[P_PROJ_W], P[P_PROJ_B], nullptr, 0, XE, 256, 256, 128, 0);
    k_split<<<(unsigned)((szC + 255) / 256), 256, 0, stream>>>(XE, SA, FA, M);
    // s0 = gelu(conv3x3(x_s))
    k_conv3x3_gelu<<<(unsigned)M, 128, 0, stream>>>(SA, P[P_SIC_W], P[P_SIC_B], SB, B, H, W, C);
    // x_f = dct2d(x_f); f0 = gelu(conv1x1(x_f))
    k_dct2d<<<B * C, 256, 0, stream>>>(FA, FB, Dm, C, 0);
    gemm(FB, 128, P[P_FIC_W], P[P_FIC_B], nullptr, 0, FA, 128, 128, 128, 1);

    // --- stage 1: dense multi-range attention ---
    gnorm(SB, P[P_SN1_G], P[P_SN1_B], SB);
    qkv(SB, SB, P_SDA); zero_acc();
    winattn(Qb, Kb, Vb, ACC, 7, 1, 0, 1.f / 3.f);
    winattn(Qb, Kb, Vb, ACC, 9, 1, 0, 1.f / 3.f);
    winattn(Qb, Kb, Vb, ACC, 11, 1, 0, 1.f / 3.f);
    gemm(ACC, 128, P[P_SDA + 6], P[P_SDA + 7], SB, 128, SA, 128, 128, 128, 0);
    ffn(SA, P_SFFN1, SB);                                   // s1 -> SB

    gnorm(FA, P[P_FN1_G], P[P_FN1_B], FA);
    qkv(FA, FA, P_FDA); zero_acc();
    winattn(Qb, Kb, Vb, ACC, 7, 1, 0, 1.f / 3.f);
    winattn(Qb, Kb, Vb, ACC, 9, 1, 0, 1.f / 3.f);
    winattn(Qb, Kb, Vb, ACC, 11, 1, 0, 1.f / 3.f);
    gemm(ACC, 128, P[P_FDA + 6], P[P_FDA + 7], FA, 128, FB, 128, 128, 128, 0);
    ffn(FB, P_FFFN1, FA);                                   // f1 -> FA

    // --- stage 2: shifted cross attention ---
    gnorm(SB, P[P_SN2_G], P[P_SN2_B], SC);                  // s2 base (keep s1=SB)
    qkv(SC, FA, P_SCA1); zero_acc();
    winattn(Qb, Kb, Vb, ACC, 8, 1, 4, 1.f);
    gemm(ACC, 128, P[P_SCA1 + 6], P[P_SCA1 + 7], SC, 128, SA, 128, 128, 128, 0);
    ffn(SA, P_SFFN2, SC);                                   // s2 -> SC

    gnorm(FA, P[P_FN2_G], P[P_FN2_B], FC);                  // f2 base (keep f1=FA)
    qkv(FC, SB, P_FCA1); zero_acc();
    winattn(Qb, Kb, Vb, ACC, 8, 1, 4, 1.f);
    gemm(ACC, 128, P[P_FCA1 + 6], P[P_FCA1 + 7], FC, 128, FB, 128, 128, 128, 0);
    ffn(FB, P_FFFN2, FC);                                   // f2 -> FC

    // --- inter projections ---
    gemm(SC, 128, P[P_SINTER_W], P[P_SINTER_B], nullptr, 0, SA, 128, 128, 128, 1); // s3
    gemm(FC, 128, P[P_FINTER_W], P[P_FINTER_B], nullptr, 0, FA, 128, 128, 128, 1); // f3

    // --- stage 3: sparse multi-range attention ---
    gnorm(SA, P[P_SN3_G], P[P_SN3_B], SA);
    qkv(SA, SA, P_SSA); zero_acc();
    winattn(Qb, Kb, Vb, ACC, 3, 1, 0, 0.25f);
    winattn(Qb, Kb, Vb, ACC, 5, 2, 0, 0.25f);
    winattn(Qb, Kb, Vb, ACC, 7, 3, 0, 0.25f);
    winattn(Qb, Kb, Vb, ACC, 9, 4, 0, 0.25f);
    gemm(ACC, 128, P[P_SSA + 6], P[P_SSA + 7], SA, 128, SB, 128, 128, 128, 0);
    ffn(SB, P_SFFN3, SA);                                   // s4 -> SA

    gnorm(FA, P[P_FN3_G], P[P_FN3_B], FA);
    qkv(FA, FA, P_FSA); zero_acc();
    winattn(Qb, Kb, Vb, ACC, 3, 1, 0, 0.25f);
    winattn(Qb, Kb, Vb, ACC, 5, 2, 0, 0.25f);
    winattn(Qb, Kb, Vb, ACC, 7, 3, 0, 0.25f);
    winattn(Qb, Kb, Vb, ACC, 9, 4, 0, 0.25f);
    gemm(ACC, 128, P[P_FSA + 6], P[P_FSA + 7], FA, 128, FB, 128, 128, 128, 0);
    ffn(FB, P_FFFN3, FA);                                   // f4 -> FA

    // --- stage 4: channel cross attention ---
    gnorm(SA, P[P_SN4_G], P[P_SN4_B], SC);                  // s5 base (keep s4=SA)
    qkv(SC, FA, P_SCA2);
    k_chan_attn<<<B * 1 * (H * W / CCS), 128, 0, stream>>>(Qb, Kb, Vb, ACC, B, C, H * W, 1);
    gemm(ACC, 128, P[P_SCA2 + 6], P[P_SCA2 + 7], SC, 128, SB, 128, 128, 128, 0);
    ffn(SB, P_SFFN4, SC);                                   // s5 -> SC

    gnorm(FA, P[P_FN4_G], P[P_FN4_B], FC);                  // f5 base (keep f4=FA)
    qkv(FC, SA, P_FCA2);
    k_chan_attn<<<B * 4 * (H * W / CCS), 128, 0, stream>>>(Qb, Kb, Vb, ACC, B, C, H * W, 4);
    gemm(ACC, 128, P[P_FCA2 + 6], P[P_FCA2 + 7], FC, 128, FB, 128, 128, 128, 0);
    ffn(FB, P_FFFN4, FC);                                   // f5 -> FC

    // --- output head ---
    k_dct2d<<<B * C, 256, 0, stream>>>(FC, FB, Dm, C, 1);   // idct2d(f5) -> FB
    k_concat_add<<<(unsigned)((M * 256 + 255) / 256), 256, 0, stream>>>(XE, SC, FB, RSB, M);
    gemm(RSB, 256, P[P_OUT_W], P[P_OUT_B], nullptr, 0, SA, 128, 128, 256, 0);
    gnorm(SA, P[P_OUTN_G], P[P_OUTN_B], SA);
    {
        const size_t total = (size_t)B * C * H * W;
        k_tok_to_nchw<<<(unsigned)((total + 255) / 256), 256, 0, stream>>>(SA, (float*)d_out, B, C, H, W);
    }
}
